// SegFormerMultiheadAttention_36464272343620
// MI455X (gfx1250) — matrix operand order repaired and checked by reading, not compiled, not run on hardware
//
#include <hip/hip_runtime.h>
#include <hip/hip_bf16.h>
#include <stdint.h>

// ---- CDNA5 WMMA types ------------------------------------------------------
typedef __attribute__((ext_vector_type(16))) __bf16 v16bf;
typedef __attribute__((ext_vector_type(8)))  float  v8f;
typedef __attribute__((ext_vector_type(4)))  unsigned int v4u;
typedef __attribute__((ext_vector_type(8)))  int v8i_;
typedef __attribute__((ext_vector_type(4)))  int v4i_;

#define WMMA_BF16(a, b, c) \
  __builtin_amdgcn_wmma_f32_16x16x32_bf16(false, (a), false, (b), (short)0, (c), false, false)

__device__ __forceinline__ __bf16 f2bf(float f) {
#if defined(__gfx1250__) && __has_builtin(__builtin_amdgcn_cvt_pk_bf16_f32)
  auto p = __builtin_amdgcn_cvt_pk_bf16_f32(f, 0.f);   // low element = cvt(f)
  __bf16 b;
  __builtin_memcpy(&b, &p, sizeof(b));
  return b;
#else
  union { float f; uint32_t u; } in; in.f = f;
  uint32_t u = in.u;
  u += 0x7fffu + ((u >> 16) & 1u);              // round-to-nearest-even
  union { uint16_t u; __bf16 b; } out; out.u = (uint16_t)(u >> 16);
  return out.b;
#endif
}

// A-fragment K index for 16-bit 16x32 A: lane holds row M=lane&15;
// element e -> K = e + 16*(lane>=16)
__device__ __forceinline__ int kA(int e, int hi) {
  return e + (hi ? 16 : 0);
}

// ---- TDM: 1-D bulk load global -> LDS (CDNA5 Tensor Data Mover) ------------
// n8 = number of 8-byte elements (fits in 16-bit tile_dim0).
// This toolchain exposes the 6-arg builtin: (v4u, v8i, v4i, v4i, v8i, cpol).
__device__ __forceinline__ void tdm_load_1d(unsigned lds_addr, const void* gptr, unsigned n8) {
#if defined(__gfx1250__) && __has_builtin(__builtin_amdgcn_tensor_load_to_lds)
  unsigned long long ga = (unsigned long long)(uintptr_t)gptr;
  // D# group 0: [1:0]=count=1, lds_addr, 57-bit global addr, type=2 (bits 127:126)
  v4u g0 = { 1u,
             lds_addr,
             (unsigned)ga,
             (unsigned)((ga >> 32) & 0x1FFFFFFu) | (2u << 30) };
  // D# group 1: data_size=3 (8B); tensor_dim0=n8 (bits 79:48); tensor_dim1=1
  // (bits 111:80); tile_dim0=n8 (bits 127:112); tile_dim1/2=0 (1-D);
  // tensor_dim0_stride=n8 (bits 207:160).
  v8i_ g1 = { (int)(3u << 16),
              (int)((n8 & 0xFFFFu) << 16),
              (int)(((n8 >> 16) & 0xFFFFu) | (1u << 16)),
              (int)((n8 & 0xFFFFu) << 16),
              0,
              (int)n8,
              0, 0 };
  v4i_ z4 = { 0, 0, 0, 0 };
  v8i_ z8 = { 0, 0, 0, 0, 0, 0, 0, 0 };
  __builtin_amdgcn_tensor_load_to_lds(g0, g1, z4, z4, z8, 0);
#endif
}

__device__ __forceinline__ void tdm_wait() {
#if defined(__gfx1250__) && __has_builtin(__builtin_amdgcn_s_wait_tensorcnt)
  __builtin_amdgcn_s_wait_tensorcnt(0);
#endif
}

// ---- problem constants -----------------------------------------------------
// B=8, N=16384 (H=W=128), C=64, SR=8 -> Nr=256, heads=1, d=64, scale=0.125

// ============================================================================
// Kernel 0: convert weights to bf16, transposed to N-major / K-contiguous so
// B fragments are contiguous 32B loads.
// ============================================================================
__global__ void kconv_weights(const float* __restrict__ Wq, const float* __restrict__ Wk,
                              const float* __restrict__ Wv, const float* __restrict__ Wp,
                              const float* __restrict__ Wsr,
                              unsigned short* __restrict__ Wqt_, unsigned short* __restrict__ Wkt_,
                              unsigned short* __restrict__ Wvt_, unsigned short* __restrict__ Wpt_,
                              unsigned short* __restrict__ Wsrt_) {
  __bf16* Wqt  = (__bf16*)Wqt_;  __bf16* Wkt = (__bf16*)Wkt_;
  __bf16* Wvt  = (__bf16*)Wvt_;  __bf16* Wpt = (__bf16*)Wpt_;
  __bf16* Wsrt = (__bf16*)Wsrt_;
  int idx = blockIdx.x * blockDim.x + threadIdx.x;
  int stride = gridDim.x * blockDim.x;
  for (int i = idx; i < 64 * 64; i += stride) {
    int k = i >> 6, n = i & 63;
    Wqt[n * 64 + k] = f2bf(Wq[i]);
    Wkt[n * 64 + k] = f2bf(Wk[i]);
    Wvt[n * 64 + k] = f2bf(Wv[i]);
    Wpt[n * 64 + k] = f2bf(Wp[i]);
  }
  for (int i = idx; i < 4096 * 64; i += stride) {
    int k = i >> 6, n = i & 63;
    Wsrt[(size_t)n * 4096 + k] = f2bf(Wsr[i]);
  }
}

// ============================================================================
// Kernel 1: sequence reduction (8x8 strided conv == 2048x4096x64 GEMM) +
// bias + LayerNorm + K/V projections. One wave per 16 output rows.
// ============================================================================
__global__ void __launch_bounds__(128)
kreduce_kv(const float* __restrict__ x, const float* __restrict__ bsr,
           const float* __restrict__ gamma, const float* __restrict__ beta,
           const float* __restrict__ bk, const float* __restrict__ bv,
           const unsigned short* __restrict__ Wsrt_, const unsigned short* __restrict__ Wkt_,
           const unsigned short* __restrict__ Wvt_,
           unsigned short* __restrict__ Kbf_, unsigned short* __restrict__ Vbf_) {
  const __bf16* Wsrt = (const __bf16*)Wsrt_;
  const __bf16* Wkt  = (const __bf16*)Wkt_;
  const __bf16* Wvt  = (const __bf16*)Wvt_;
  __bf16* Kbf = (__bf16*)Kbf_;
  __bf16* Vbf = (__bf16*)Vbf_;

  __shared__ __bf16 xst[4][16 * 64];

  int tid = threadIdx.x;
  int wave = tid >> 5, lane = tid & 31;
  int hi = lane >> 4, col = lane & 15;
  int t = blockIdx.x * 4 + wave;          // M-tile index, 0..127
  int prow = t * 16 + col;                // this lane's A row (patch index)
  int bb = prow >> 8;                     // batch
  int pi = (prow >> 4) & 15, pj = prow & 15;
  const float* xb = x + (size_t)bb * (16384 * 64);

  v8f acc[4];
  { v8f z = {}; acc[0] = z; acc[1] = z; acc[2] = z; acc[3] = z; }

  // K = 4096 contraction over the 8x8x64 patch, 32 per WMMA step
  #pragma unroll 1
  for (int ks = 0; ks < 128; ++ks) {
    // prefetch this lane's gather stream a few k-steps ahead
    if (ks + 4 < 128) {
      int kp = (ks + 4) * 32 + kA(0, hi);
      int kip = kp >> 9, kjp = (kp >> 6) & 7, cinp = kp & 63;
      __builtin_prefetch(&xb[(((pi * 8 + kip) * 128) + (pj * 8 + kjp)) * 64 + cinp], 0, 0);
    }
    v16bf a;
    #pragma unroll
    for (int e = 0; e < 16; ++e) {
      int k = ks * 32 + kA(e, hi);
      int ki = k >> 9, kj = (k >> 6) & 7, cin = k & 63;
      a[e] = f2bf(xb[(((pi * 8 + ki) * 128) + (pj * 8 + kj)) * 64 + cin]);
    }
    #pragma unroll
    for (int nt = 0; nt < 4; ++nt) {
      v16bf wb = *(const v16bf*)(Wsrt + (size_t)(nt * 16 + col) * 4096 + ks * 32 + hi * 16);
      acc[nt] = WMMA_BF16(a, wb, acc[nt]);
    }
  }

  // bias + LayerNorm over C=64 (rows r+8*hi live in this lane's half)
  float mean[8], rstd[8];
  {
    float s1[8], s2[8];
    #pragma unroll
    for (int r = 0; r < 8; ++r) { s1[r] = 0.f; s2[r] = 0.f; }
    #pragma unroll
    for (int nt = 0; nt < 4; ++nt) {
      float bs = bsr[nt * 16 + col];
      #pragma unroll
      for (int r = 0; r < 8; ++r) {
        float v = acc[nt][r] + bs;
        acc[nt][r] = v;
        s1[r] += v; s2[r] += v * v;
      }
    }
    #pragma unroll
    for (int r = 0; r < 8; ++r) {
      float a1 = s1[r], a2 = s2[r];
      for (int m = 1; m < 16; m <<= 1) {
        a1 += __shfl_xor(a1, m, 32);
        a2 += __shfl_xor(a2, m, 32);
      }
      float mu = a1 * (1.f / 64.f);
      float var = a2 * (1.f / 64.f) - mu * mu;
      mean[r] = mu;
      rstd[r] = rsqrtf(var + 1e-5f);
    }
  }
  #pragma unroll
  for (int nt = 0; nt < 4; ++nt) {
    float g = gamma[nt * 16 + col], be = beta[nt * 16 + col];
    #pragma unroll
    for (int r = 0; r < 8; ++r) {
      float v = (acc[nt][r] - mean[r]) * rstd[r] * g + be;
      xst[wave][(r + 8 * hi) * 64 + nt * 16 + col] = f2bf(v);
    }
  }
  __syncthreads();

  // re-layout D -> A fragments through LDS
  v16bf xa[2];
  #pragma unroll
  for (int kc = 0; kc < 2; ++kc)
    #pragma unroll
    for (int e = 0; e < 16; ++e)
      xa[kc][e] = xst[wave][col * 64 + kc * 32 + kA(e, hi)];

  // K and V projections (64x64), write bf16 to workspace
  #pragma unroll
  for (int nt = 0; nt < 4; ++nt) {
    v8f kacc = {}; v8f vacc = {};
    #pragma unroll
    for (int kc = 0; kc < 2; ++kc) {
      v16bf wbk = *(const v16bf*)(Wkt + (nt * 16 + col) * 64 + kc * 32 + hi * 16);
      v16bf wbv = *(const v16bf*)(Wvt + (nt * 16 + col) * 64 + kc * 32 + hi * 16);
      kacc = WMMA_BF16(xa[kc], wbk, kacc);
      vacc = WMMA_BF16(xa[kc], wbv, vacc);
    }
    float bkc = bk[nt * 16 + col], bvc = bv[nt * 16 + col];
    #pragma unroll
    for (int r = 0; r < 8; ++r) {
      int row = t * 16 + r + 8 * hi;
      Kbf[(size_t)row * 64 + nt * 16 + col] = f2bf(kacc[r] + bkc);
      Vbf[(size_t)row * 64 + nt * 16 + col] = f2bf(vacc[r] + bvc);
    }
  }
}

// ============================================================================
// Kernel 2: fused Q-proj + attention (256 keys, LDS-resident K/V) + out-proj.
// Workgroup = 4 waves x 16 query rows = 64 rows; grid = 8 batches * 256.
// K/V brought into LDS by the Tensor Data Mover (TENSORcnt), one 8KB quarter
// per wave for each of K and V.
// ============================================================================
__global__ void __launch_bounds__(128)
kattn(const float* __restrict__ x, const float* __restrict__ bq, const float* __restrict__ bp,
      const unsigned short* __restrict__ Wqt_, const unsigned short* __restrict__ Wpt_,
      const unsigned short* __restrict__ Kbf_, const unsigned short* __restrict__ Vbf_,
      float* __restrict__ out) {
  const __bf16* Wqt = (const __bf16*)Wqt_;
  const __bf16* Wpt = (const __bf16*)Wpt_;
  const __bf16* Kbf = (const __bf16*)Kbf_;
  const __bf16* Vbf = (const __bf16*)Vbf_;

  __shared__ __bf16 Klds[256 * 64];      // 32 KB
  __shared__ __bf16 Vlds[256 * 64];      // 32 KB
  __shared__ __bf16 stage[4][16 * 64];   // 8 KB  (q / o re-layout)
  __shared__ __bf16 Pst[4][16 * 256];    // 32 KB (P re-layout)

  int tid = threadIdx.x;
  int wave = tid >> 5, lane = tid & 31;
  int hi = lane >> 4, col = lane & 15;
  int b = blockIdx.x >> 8;
  int mbase = (blockIdx.x & 255) * 64 + wave * 16;

#if defined(__gfx1250__) && __has_builtin(__builtin_amdgcn_tensor_load_to_lds)
  // ---- TDM async fill: each wave moves its 8KB quarter of K and of V ----
  {
    unsigned koff = (unsigned)(wave * 8192);
    unsigned klds = (unsigned)(uintptr_t)&Klds[0] + koff;   // low 32 bits = LDS byte addr
    unsigned vlds = (unsigned)(uintptr_t)&Vlds[0] + koff;
    const char* kg = (const char*)(Kbf + (size_t)b * 256 * 64) + koff;
    const char* vg = (const char*)(Vbf + (size_t)b * 256 * 64) + koff;
    tdm_load_1d(klds, kg, 1024);   // 1024 x 8B = 8KB
    tdm_load_1d(vlds, vg, 1024);
    tdm_wait();
  }
#else
  // fallback: cooperative vector copy
  {
    const uint4* ksrc = (const uint4*)(Kbf + (size_t)b * 256 * 64);
    const uint4* vsrc = (const uint4*)(Vbf + (size_t)b * 256 * 64);
    uint4* kdst = (uint4*)Klds;
    uint4* vdst = (uint4*)Vlds;
    for (int i = tid; i < 2048; i += 128) { kdst[i] = ksrc[i]; vdst[i] = vsrc[i]; }
  }
#endif
  __syncthreads();

  // ---- Q projection for this wave's 16 rows (scale folded in) ----
  {
    const float* xrow = x + ((size_t)b * 16384 + mbase + col) * 64;
    v16bf xa[2];
    #pragma unroll
    for (int kc = 0; kc < 2; ++kc)
      #pragma unroll
      for (int e = 0; e < 16; ++e)
        xa[kc][e] = f2bf(xrow[kc * 32 + kA(e, hi)]);
    #pragma unroll
    for (int nt = 0; nt < 4; ++nt) {
      v8f qd = {};
      #pragma unroll
      for (int kc = 0; kc < 2; ++kc) {
        v16bf wb = *(const v16bf*)(Wqt + (nt * 16 + col) * 64 + kc * 32 + hi * 16);
        qd = WMMA_BF16(xa[kc], wb, qd);
      }
      float bqc = bq[nt * 16 + col];
      #pragma unroll
      for (int r = 0; r < 8; ++r)
        stage[wave][(r + 8 * hi) * 64 + nt * 16 + col] = f2bf((qd[r] + bqc) * 0.125f);
    }
  }
  __syncthreads();

  v16bf qa[2];
  #pragma unroll
  for (int kc = 0; kc < 2; ++kc)
    #pragma unroll
    for (int e = 0; e < 16; ++e)
      qa[kc][e] = stage[wave][col * 64 + kc * 32 + kA(e, hi)];

  // ---- scores S = q @ K^T : 16 tiles of 16x16 ----
  v8f s[16];
  #pragma unroll
  for (int nt = 0; nt < 16; ++nt) {
    v8f a = {};
    #pragma unroll
    for (int kc = 0; kc < 2; ++kc) {
      v16bf kb = *(const v16bf*)(&Klds[(nt * 16 + col) * 64 + kc * 32 + hi * 16]);
      a = WMMA_BF16(qa[kc], kb, a);
    }
    s[nt] = a;
  }

  // ---- softmax over 256 keys (rows r+8*hi within this lane half) ----
  float rmax[8], rsum[8];
  #pragma unroll
  for (int r = 0; r < 8; ++r) {
    float m = -3.4e38f;
    #pragma unroll
    for (int nt = 0; nt < 16; ++nt) m = fmaxf(m, s[nt][r]);
    for (int msk = 1; msk < 16; msk <<= 1) m = fmaxf(m, __shfl_xor(m, msk, 32));
    rmax[r] = m;
    rsum[r] = 0.f;
  }
  #pragma unroll
  for (int nt = 0; nt < 16; ++nt)
    #pragma unroll
    for (int r = 0; r < 8; ++r) {
      float p = __expf(s[nt][r] - rmax[r]);
      s[nt][r] = p;
      rsum[r] += p;
    }
  #pragma unroll
  for (int r = 0; r < 8; ++r) {
    float a1 = rsum[r];
    for (int msk = 1; msk < 16; msk <<= 1) a1 += __shfl_xor(a1, msk, 32);
    rsum[r] = a1;
  }

  // stage P (D layout -> row-major bf16) for A-fragment reload
  #pragma unroll
  for (int nt = 0; nt < 16; ++nt)
    #pragma unroll
    for (int r = 0; r < 8; ++r)
      Pst[wave][(r + 8 * hi) * 256 + nt * 16 + col] = f2bf(s[nt][r]);
  __syncthreads();

  // ---- O = P @ V (contraction over 256 keys, 8 WMMA k-steps) ----
  v8f od[4];
  { v8f z = {}; od[0] = z; od[1] = z; od[2] = z; od[3] = z; }
  #pragma unroll 1
  for (int ksn = 0; ksn < 8; ++ksn) {
    v16bf pa;
    #pragma unroll
    for (int e = 0; e < 16; ++e)
      pa[e] = Pst[wave][col * 256 + ksn * 32 + kA(e, hi)];
    #pragma unroll
    for (int dt = 0; dt < 4; ++dt) {
      v16bf vb;
      #pragma unroll
      for (int e = 0; e < 16; ++e)
        vb[e] = Vlds[(ksn * 32 + hi * 16 + e) * 64 + dt * 16 + col];
      od[dt] = WMMA_BF16(pa, vb, od[dt]);
    }
  }

  // normalize and stage O for the fused output projection
  #pragma unroll
  for (int dt = 0; dt < 4; ++dt)
    #pragma unroll
    for (int r = 0; r < 8; ++r)
      stage[wave][(r + 8 * hi) * 64 + dt * 16 + col] = f2bf(od[dt][r] / rsum[r]);
  __syncthreads();

  // ---- final = O @ Wp + bp ----
  v16bf oa[2];
  #pragma unroll
  for (int kc = 0; kc < 2; ++kc)
    #pragma unroll
    for (int e = 0; e < 16; ++e)
      oa[kc][e] = stage[wave][col * 64 + kc * 32 + kA(e, hi)];
  #pragma unroll
  for (int nt = 0; nt < 4; ++nt) {
    v8f fd = {};
    #pragma unroll
    for (int kc = 0; kc < 2; ++kc) {
      v16bf wb = *(const v16bf*)(Wpt + (nt * 16 + col) * 64 + kc * 32 + hi * 16);
      fd = WMMA_BF16(oa[kc], wb, fd);
    }
    float bpc = bp[nt * 16 + col];
    #pragma unroll
    for (int r = 0; r < 8; ++r)
      out[((size_t)b * 16384 + mbase + r + 8 * hi) * 64 + nt * 16 + col] = fd[r] + bpc;
  }
}

// ============================================================================
extern "C" void kernel_launch(void* const* d_in, const int* in_sizes, int n_in,
                              void* d_out, int out_size, void* d_ws, size_t ws_size,
                              hipStream_t stream) {
  const float* x     = (const float*)d_in[0];
  const float* Wq    = (const float*)d_in[1];
  const float* bq    = (const float*)d_in[2];
  const float* Wk    = (const float*)d_in[3];
  const float* bk    = (const float*)d_in[4];
  const float* Wv    = (const float*)d_in[5];
  const float* bv    = (const float*)d_in[6];
  const float* Wsr   = (const float*)d_in[7];
  const float* bsr   = (const float*)d_in[8];
  const float* gamma = (const float*)d_in[9];
  const float* beta  = (const float*)d_in[10];
  const float* Wp    = (const float*)d_in[11];
  const float* bp    = (const float*)d_in[12];
  float* out = (float*)d_out;

  char* ws = (char*)d_ws;
  unsigned short* Wqt  = (unsigned short*)(ws + 0);        //  8 KB
  unsigned short* Wkt  = (unsigned short*)(ws + 8192);     //  8 KB
  unsigned short* Wvt  = (unsigned short*)(ws + 16384);    //  8 KB
  unsigned short* Wpt  = (unsigned short*)(ws + 24576);    //  8 KB
  unsigned short* Wsrt = (unsigned short*)(ws + 32768);    // 512 KB
  unsigned short* Kbf  = (unsigned short*)(ws + 557056);   // 256 KB
  unsigned short* Vbf  = (unsigned short*)(ws + 819200);   // 256 KB

  hipLaunchKernelGGL(kconv_weights, dim3(64), dim3(256), 0, stream,
                     Wq, Wk, Wv, Wp, Wsr, Wqt, Wkt, Wvt, Wpt, Wsrt);
  hipLaunchKernelGGL(kreduce_kv, dim3(32), dim3(128), 0, stream,
                     x, bsr, gamma, beta, bk, bv, Wsrt, Wkt, Wvt, Kbf, Vbf);
  hipLaunchKernelGGL(kattn, dim3(2048), dim3(128), 0, stream,
                     x, bq, bp, Wqt, Wpt, Kbf, Vbf, out);
}
